// SparseVariationalAttention_13443247637236
// MI455X (gfx1250) — matrix operand order repaired and checked by reading, not compiled, not run on hardware
//
#include <hip/hip_runtime.h>
#include <hip/hip_bf16.h>
#include <math.h>

// ---------------- WMMA types ----------------
typedef __attribute__((ext_vector_type(16))) _Float16 v16h;
typedef __attribute__((ext_vector_type(8)))  _Float16 v8h;
typedef __attribute__((ext_vector_type(8)))  float    v8f;

#define BM 128
#define BN 64
#define BK 32
#define T_DIM 2048
#define C_DIM 512
#define H_DIM 8
#define HD_DIM 64
#define U_TOP 40
#define NCHUNK 16
#define TC 128
#define SIGMA_F 0.69314718055994531f

// ============================================================
// K1/K6: Y = A @ W^T + bias  (M x K) x (N x K)^T, WMMA f16->f32
// MODE 0: write as q/k/v layout (B,H,T,HD); MODE 1: plain row-major
// ============================================================
template<int MODE>
__global__ __launch_bounds__(256)
void gemm_wmma_kernel(const float* __restrict__ A,
                      const float* __restrict__ W0, const float* __restrict__ W1,
                      const float* __restrict__ W2,
                      const float* __restrict__ bias0, const float* __restrict__ bias1,
                      const float* __restrict__ bias2,
                      float* o0, float* o1, float* o2,
                      int Kdim, int Ncols)
{
  const int z = blockIdx.z;
  const float* W    = (z == 0) ? W0 : ((z == 1) ? W1 : W2);
  const float* bias = (z == 0) ? bias0 : ((z == 1) ? bias1 : bias2);
  float* out        = (z == 0) ? o0 : ((z == 1) ? o1 : o2);

  __shared__ __align__(32) _Float16 As[BM * 48];   // 48-halfword row stride: 96B (32B aligned)
  __shared__ __align__(32) _Float16 Bs[BN * 48];

  const int tid   = threadIdx.x;
  const int lane  = tid & 31;
  const int wid   = tid >> 5;        // 8 waves
  const int waveM = wid >> 1;        // 0..3
  const int waveN = wid & 1;         // 0..1
  const int half  = lane >> 4;
  const int lm    = lane & 15;

  const int rowBase = blockIdx.y * BM;
  const int colBase = blockIdx.x * BN;

  v8f c[2][2] = {};

  const int ar  = tid & 127;         // A tile row
  const int acg = tid >> 7;          // 16-col group
  const int br  = tid & 63;          // B tile row
  const int bcg = (tid >> 6) & 1;

  for (int k0 = 0; k0 < Kdim; k0 += BK) {
    { // A tile: fp32 -> f16
      const float* src = A + (size_t)(rowBase + ar) * Kdim + k0 + acg * 16;
      _Float16* dst = &As[ar * 48 + acg * 16];
#pragma unroll
      for (int j = 0; j < 4; ++j) {
        float4 f = ((const float4*)src)[j];
        dst[4*j+0] = (_Float16)f.x; dst[4*j+1] = (_Float16)f.y;
        dst[4*j+2] = (_Float16)f.z; dst[4*j+3] = (_Float16)f.w;
      }
      if (k0 + BK < Kdim) __builtin_prefetch(src + BK, 0, 0);  // global_prefetch_b8
    }
    if (tid < 128) { // B tile: fp32 -> f16
      const float* src = W + (size_t)(colBase + br) * Kdim + k0 + bcg * 16;
      _Float16* dst = &Bs[br * 48 + bcg * 16];
#pragma unroll
      for (int j = 0; j < 4; ++j) {
        float4 f = ((const float4*)src)[j];
        dst[4*j+0] = (_Float16)f.x; dst[4*j+1] = (_Float16)f.y;
        dst[4*j+2] = (_Float16)f.z; dst[4*j+3] = (_Float16)f.w;
      }
    }
    __syncthreads();

    v16h afrag[2], bfrag[2];
#pragma unroll
    for (int mi = 0; mi < 2; ++mi) {
      int m = waveM * 32 + mi * 16 + lm;
      v8h lo = *(const v8h*)(As + m * 48 + half * 16);       // K = half*16 + 0..7
      v8h hi = *(const v8h*)(As + m * 48 + half * 16 + 8);   // K = half*16 + 8..15
#pragma unroll
      for (int j = 0; j < 8; ++j) { afrag[mi][j] = lo[j]; afrag[mi][8 + j] = hi[j]; }
    }
#pragma unroll
    for (int ni = 0; ni < 2; ++ni) {
      int n = waveN * 32 + ni * 16 + lm;
      bfrag[ni] = *(const v16h*)(Bs + n * 48 + half * 16);   // K = half*16 + 0..15
    }
#pragma unroll
    for (int mi = 0; mi < 2; ++mi)
#pragma unroll
      for (int ni = 0; ni < 2; ++ni)
        c[mi][ni] = __builtin_amdgcn_wmma_f32_16x16x32_f16(
            false, afrag[mi], false, bfrag[ni], (short)0, c[mi][ni], false, false);
    __syncthreads();
  }

  // epilogue: C/D layout: lane -> N = lm, VGPR r -> M = r + 8*half
#pragma unroll
  for (int mi = 0; mi < 2; ++mi) {
#pragma unroll
    for (int ni = 0; ni < 2; ++ni) {
      int ng = colBase + waveN * 32 + ni * 16 + lm;
      float bv = bias[ng];
#pragma unroll
      for (int r = 0; r < 8; ++r) {
        int mg = rowBase + waveM * 32 + mi * 16 + r + 8 * half;
        float val = c[mi][ni][r] + bv;
        if (MODE == 0) {
          int bb = mg >> 11;          // / T
          int tt = mg & (T_DIM - 1);
          int hh = ng >> 6;           // / HD
          int hd = ng & (HD_DIM - 1);
          out[((((size_t)bb * H_DIM + hh) * T_DIM) + tt) * HD_DIM + hd] = val;
        } else {
          out[(size_t)mg * Ncols + ng] = val;
        }
      }
    }
  }
}

// ============================================================
// K2: sparsity measure  M = max_u(q . k_sample) - mean_u(...)
// ============================================================
__global__ __launch_bounds__(256)
void sample_m_kernel(const float* __restrict__ q, const float* __restrict__ k,
                     const int* __restrict__ isamp, float* __restrict__ Mout)
{
  int g  = blockIdx.x * 256 + threadIdx.x;   // (b*H+h)*T + t
  int t  = g & (T_DIM - 1);
  int bh = g >> 11;
  const float4* qr = (const float4*)(q + ((size_t)bh * T_DIM + t) * HD_DIM);
  float4 qv[16];
#pragma unroll
  for (int i = 0; i < 16; ++i) qv[i] = qr[i];
  float mx = -INFINITY, sm = 0.f;
  for (int u = 0; u < U_TOP; ++u) {
    int j = isamp[t * U_TOP + u];
    const float4* kr = (const float4*)(k + ((size_t)bh * T_DIM + j) * HD_DIM);
    float d = 0.f;
#pragma unroll
    for (int i = 0; i < 16; ++i) {
      float4 kv = kr[i];
      d += qv[i].x * kv.x + qv[i].y * kv.y + qv[i].z * kv.z + qv[i].w * kv.w;
    }
    mx = fmaxf(mx, d);
    sm += d;
  }
  Mout[g] = mx - sm * (1.0f / (float)U_TOP);
}

// ============================================================
// K3: per-(b,h) top-40 by iterative argmax (ties -> smaller idx)
// ============================================================
__global__ __launch_bounds__(256)
void topk_kernel(const float* __restrict__ Mbuf, int* __restrict__ topidx)
{
  __shared__ float ml[T_DIM];
  __shared__ float rv[256];
  __shared__ int   ri[256];
  int bh = blockIdx.x, tid = threadIdx.x;
  for (int i = tid; i < T_DIM; i += 256) ml[i] = Mbuf[(size_t)bh * T_DIM + i];
  __syncthreads();
  for (int it = 0; it < U_TOP; ++it) {
    float bv = -INFINITY; int bi = 0x7fffffff;
    for (int i = tid; i < T_DIM; i += 256) {
      float x = ml[i];
      if (x > bv) { bv = x; bi = i; }
    }
    rv[tid] = bv; ri[tid] = bi;
    __syncthreads();
    for (int s = 128; s > 0; s >>= 1) {
      if (tid < s) {
        float ov = rv[tid + s]; int oi = ri[tid + s];
        if (ov > rv[tid] || (ov == rv[tid] && oi < ri[tid])) { rv[tid] = ov; ri[tid] = oi; }
      }
      __syncthreads();
    }
    if (tid == 0) { topidx[bh * U_TOP + it] = ri[0]; ml[ri[0]] = -INFINITY; }
    __syncthreads();
  }
}

// ============================================================
// K4: split-T flash attention over the 40 selected rows
//   scores = Qr.K^T/8, kl += s^2, sampled = s + sigma*noise,
//   online-softmax chunk partials (m, l, o[64]) per (b,h,u,chunk)
// ============================================================
__global__ __launch_bounds__(128)
void attn_partial_kernel(const float* __restrict__ q, const float* __restrict__ k,
                         const float* __restrict__ v, const float* __restrict__ noise,
                         const int* __restrict__ topidx,
                         float* __restrict__ pm, float* __restrict__ pl,
                         float* __restrict__ po, float* __restrict__ klsum)
{
  __shared__ float ks[TC * 66];     // 66-stride: even (b64-safe) + conflict-free
  __shared__ float vs[TC * 66];
  __shared__ float qs[8 * HD_DIM];
  __shared__ float pb[TC];
  __shared__ float red[128];
  __shared__ float part[4 * HD_DIM];
  __shared__ int   tloc[8];

  const int tid   = threadIdx.x;
  const int bh    = blockIdx.x;     // 0..31
  const int chunk = blockIdx.y;     // 0..15
  const int ug    = blockIdx.z;     // 0..4
  const int t0    = chunk * TC;
  const size_t base = (size_t)bh * T_DIM;

  if (tid < 8) tloc[tid] = topidx[bh * U_TOP + ug * 8 + tid];
  __syncthreads();
  for (int e = tid; e < TC * HD_DIM; e += 128) {
    int r = e >> 6, cc = e & 63;
    ks[r * 66 + cc] = k[(base + t0 + r) * HD_DIM + cc];
    vs[r * 66 + cc] = v[(base + t0 + r) * HD_DIM + cc];
  }
  for (int i = tid; i < 8 * HD_DIM; i += 128) {
    int uu = i >> 6, cc = i & 63;
    qs[i] = q[(base + tloc[uu]) * HD_DIM + cc];
  }
  __syncthreads();

  float klacc = 0.f;
  const int wv = tid >> 5, lane = tid & 31;

  for (int uu = 0; uu < 8; ++uu) {
    int u = ug * 8 + uu;
    float s = 0.f;
    const float* kr = &ks[tid * 66];
    const float* qr = &qs[uu * HD_DIM];
#pragma unroll
    for (int j = 0; j < HD_DIM; ++j) s += qr[j] * kr[j];
    s *= 0.125f;                        // 1/sqrt(64)
    klacc += s * s;
    float nzv  = noise[(((size_t)bh * U_TOP + u) * T_DIM) + t0 + tid];
    float samp = s + SIGMA_F * nzv;

    red[tid] = samp; __syncthreads();
    for (int st = 64; st > 0; st >>= 1) {
      if (tid < st) red[tid] = fmaxf(red[tid], red[tid + st]);
      __syncthreads();
    }
    float mc = red[0]; __syncthreads();
    float p = __expf(samp - mc);
    pb[tid] = p;
    red[tid] = p; __syncthreads();
    for (int st = 64; st > 0; st >>= 1) {
      if (tid < st) red[tid] += red[tid + st];
      __syncthreads();
    }
    float lc = red[0];

    // context partial: wave wv handles t in [wv*32, wv*32+32), lane -> hd pair
    float a0 = 0.f, a1 = 0.f;
#pragma unroll
    for (int ttj = 0; ttj < 32; ++ttj) {
      int tt = wv * 32 + ttj;
      float pt = pb[tt];
      a0 += pt * vs[tt * 66 + 2 * lane];
      a1 += pt * vs[tt * 66 + 2 * lane + 1];
    }
    part[wv * 64 + 2 * lane]     = a0;
    part[wv * 64 + 2 * lane + 1] = a1;
    __syncthreads();
    size_t pidx = ((size_t)bh * U_TOP + u) * NCHUNK + chunk;
    if (tid < 64) {
      float o = part[tid] + part[64 + tid] + part[128 + tid] + part[192 + tid];
      po[pidx * HD_DIM + tid] = o;
    }
    if (tid == 0) { pm[pidx] = mc; pl[pidx] = lc; }
    __syncthreads();
  }

  red[tid] = klacc; __syncthreads();
  for (int st = 64; st > 0; st >>= 1) {
    if (tid < st) red[tid] += red[tid + st];
    __syncthreads();
  }
  if (tid == 0) atomicAdd(klsum, red[0]);   // global_atomic_add_f32
}

// ============================================================
// K5a: ctx_mean = mean_t v   (per b,h -> 64 values)
// ============================================================
__global__ __launch_bounds__(256)
void ctx_mean_kernel(const float* __restrict__ v, float* __restrict__ cm)
{
  __shared__ float part[4 * HD_DIM];
  int bh = blockIdx.x, tid = threadIdx.x;
  int hd = tid & 63, cg = tid >> 6;
  float s = 0.f;
  for (int t = cg * 512; t < (cg + 1) * 512; ++t)
    s += v[((size_t)bh * T_DIM + t) * HD_DIM + hd];
  part[cg * 64 + hd] = s;
  __syncthreads();
  if (tid < 64) {
    float m = (part[tid] + part[64 + tid] + part[128 + tid] + part[192 + tid]) * (1.f / (float)T_DIM);
    cm[bh * HD_DIM + tid] = m;
  }
}

// ============================================================
// K5b1: final[b,t,c] = ctx_mean broadcast  (float4 per thread)
// ============================================================
__global__ __launch_bounds__(256)
void fill_ctx_kernel(const float* __restrict__ cm, float* __restrict__ fin)
{
  size_t e4 = (size_t)blockIdx.x * 256 + threadIdx.x;
  size_t flat = e4 * 4;
  int c = (int)(flat & (C_DIM - 1));
  size_t bt = flat >> 9;
  int b = (int)(bt >> 11);
  int h = c >> 6, hd = c & 63;
  float4 mv = *(const float4*)(cm + ((size_t)b * H_DIM + h) * HD_DIM + hd);
  *(float4*)(fin + flat) = mv;
}

// ============================================================
// K5b2: combine chunk partials, normalize, scatter sparse rows
// ============================================================
__global__ __launch_bounds__(64)
void combine_kernel(const float* __restrict__ pm, const float* __restrict__ pl,
                    const float* __restrict__ po, const int* __restrict__ topidx,
                    float* __restrict__ fin)
{
  int g = blockIdx.x;                 // bh*40 + u
  int bh = g / U_TOP;
  int tid = threadIdx.x;              // hd
  size_t pbase = (size_t)g * NCHUNK;
  float m = -INFINITY;
#pragma unroll
  for (int cc = 0; cc < NCHUNK; ++cc) m = fmaxf(m, pm[pbase + cc]);
  float l = 0.f, o = 0.f;
#pragma unroll
  for (int cc = 0; cc < NCHUNK; ++cc) {
    float w = __expf(pm[pbase + cc] - m);
    l += pl[pbase + cc] * w;
    o += po[(pbase + cc) * HD_DIM + tid] * w;
  }
  float val = o / l;
  int row = topidx[g];
  int b = bh >> 3, h = bh & 7;
  fin[(((size_t)b * T_DIM) + row) * C_DIM + h * HD_DIM + tid] = val;
}

// ============================================================
// K0 / K7
// ============================================================
__global__ void zero_kernel(float* p) { p[0] = 0.f; }

__global__ void kl_final_kernel(const float* __restrict__ klsum, float* __restrict__ outkl)
{
  const float sigma = SIGMA_F;
  float cst = 0.5f * (sigma * sigma - 1.0f - 2.0f * logf(sigma + 1e-9f));
  outkl[0] = cst + 0.5f * klsum[0] * (1.0f / 2621440.0f);   // B*H*U*T
}

// ============================================================
extern "C" void kernel_launch(void* const* d_in, const int* in_sizes, int n_in,
                              void* d_out, int out_size, void* d_ws, size_t ws_size,
                              hipStream_t stream)
{
  const float* x   = (const float*)d_in[0];
  const int*   isp = (const int*)  d_in[1];
  const float* nz  = (const float*)d_in[2];
  const float* Wq  = (const float*)d_in[3];
  const float* bq  = (const float*)d_in[4];
  const float* Wk  = (const float*)d_in[5];
  const float* bk  = (const float*)d_in[6];
  const float* Wv  = (const float*)d_in[7];
  const float* bv  = (const float*)d_in[8];
  const float* Wo  = (const float*)d_in[9];
  const float* bo  = (const float*)d_in[10];
  float* out = (float*)d_out;

  // workspace layout (floats)
  float* ws = (float*)d_ws;
  const size_t QKV = (size_t)4 * T_DIM * C_DIM;      // 4,194,304 per tensor
  float* qb = ws;                                    // q; later reused as 'final'
  float* kb = qb + QKV;
  float* vb = kb + QKV;
  float* Mb = vb + QKV;                              // 65,536
  int*   ti = (int*)(Mb + 65536);                    // 1,280
  float* pm = (float*)(ti + 1280);                   // 20,480
  float* pl = pm + 20480;                            // 20,480
  float* po = pl + 20480;                            // 1,310,720
  float* cm = po + 1310720;                          // 2,048
  float* kls = cm + 2048;                            // 1

  zero_kernel<<<1, 1, 0, stream>>>(kls);

  // QKV projections (WMMA), write (B,H,T,HD)
  gemm_wmma_kernel<0><<<dim3(8, 64, 3), 256, 0, stream>>>(
      x, Wq, Wk, Wv, bq, bk, bv, qb, kb, vb, C_DIM, C_DIM);

  sample_m_kernel<<<256, 256, 0, stream>>>(qb, kb, isp, Mb);
  topk_kernel<<<32, 256, 0, stream>>>(Mb, ti);

  attn_partial_kernel<<<dim3(32, NCHUNK, 5), 128, 0, stream>>>(
      qb, kb, vb, nz, ti, pm, pl, po, kls);

  ctx_mean_kernel<<<32, 256, 0, stream>>>(vb, cm);
  fill_ctx_kernel<<<4096, 256, 0, stream>>>(cm, qb);          // final = qb (q dead now)
  combine_kernel<<<1280, 64, 0, stream>>>(pm, pl, po, ti, qb);

  // output projection (WMMA), plain row-major into d_out
  gemm_wmma_kernel<1><<<dim3(8, 64, 1), 256, 0, stream>>>(
      qb, Wo, Wo, Wo, bo, bo, bo, out, out, out, C_DIM, C_DIM);

  kl_final_kernel<<<1, 1, 0, stream>>>(kls, out + QKV);
}
